// OptimizedMoELLM_91216515432607
// MI455X (gfx1250) — matrix operand order repaired and checked by reading, not compiled, not run on hardware
//
#include <hip/hip_runtime.h>
#include <cstdint>
#include <cstddef>

// ---------------- model dims ----------------
static constexpr int B_ = 2, S_ = 2048, D_ = 1024, H_ = 16, F_ = 4096, L_ = 2, V_ = 32000;
static constexpr int HD_ = 64, IH_ = 4, ID_ = 64, TOPK_ = 64;
static constexpr int M_ = B_ * S_;              // 4096 token rows
static constexpr float NEGF = -1e30f;

typedef __bf16 bf16;
typedef __attribute__((ext_vector_type(16))) __bf16 v16bf;
typedef __attribute__((ext_vector_type(8)))  __bf16 v8bf;
typedef __attribute__((ext_vector_type(8)))  float  v8f;

union V16 { v16bf v; v8bf h[2]; };

// gfx1250 async global->LDS path (ASYNCcnt-tracked), guarded so compile never breaks
#if defined(__AMDGCN__) && __has_builtin(__builtin_amdgcn_global_load_async_to_lds_b128)
#define HAS_ASYNC 1
#else
#define HAS_ASYNC 0
#endif

// builtin params are v4i* in AS1 (global) / AS3 (LDS); cast via integers to exact types
typedef __attribute__((__vector_size__(4 * sizeof(int)))) int vsi4;
#define GPTR(p) ((__attribute__((address_space(1))) vsi4*)(uintptr_t)(p))
#define LPTR(p) ((__attribute__((address_space(3))) vsi4*)(uint32_t)(uintptr_t)(p))

__device__ __forceinline__ void wait_async0() {
#if defined(__AMDGCN__)
#if __has_builtin(__builtin_amdgcn_s_wait_asynccnt)
  __builtin_amdgcn_s_wait_asynccnt(0);
#else
  asm volatile("s_wait_asynccnt 0x0" ::: "memory");
#endif
#endif
}

__device__ __forceinline__ bf16 f2bf(float f) {
  uint32_t u = __builtin_bit_cast(uint32_t, f);
  uint32_t r = u + 0x7FFFu + ((u >> 16) & 1u);   // round-to-nearest-even
  uint16_t hh = (uint16_t)(r >> 16);
  return __builtin_bit_cast(bf16, hh);
}

// ---------------- embedding ----------------
__global__ void embed_kernel(const int* __restrict__ ids, const float* __restrict__ tok,
                             const float* __restrict__ pos, float* __restrict__ h) {
  size_t i = (size_t)blockIdx.x * blockDim.x + threadIdx.x;
  if (i >= (size_t)M_ * D_) return;
  int row = (int)(i / D_), d = (int)(i % D_);
  int s = row % S_;
  int id = ids[row];
  h[i] = tok[(size_t)id * D_ + d] + pos[(size_t)s * D_ + d];
}

// f32 [K][N] -> bf16 [N][K] (transposed weight for WMMA B operand)
__global__ void cvtT_kernel(const float* __restrict__ in, bf16* __restrict__ out, int K, int N) {
  size_t i = (size_t)blockIdx.x * blockDim.x + threadIdx.x;
  if (i >= (size_t)N * K) return;
  int n = (int)(i / K), k = (int)(i % K);
  out[i] = f2bf(in[(size_t)k * N + n]);
}

// ---------------- layernorm (row of D=1024, 256 threads) -> bf16 ----------------
__global__ __launch_bounds__(256) void ln_kernel(const float* __restrict__ x,
                                                 const float* __restrict__ g,
                                                 const float* __restrict__ bta,
                                                 bf16* __restrict__ y) {
  int row = blockIdx.x, tid = threadIdx.x;
  const float* xr = x + (size_t)row * D_;
  float v[4], s1 = 0.f, s2 = 0.f;
  for (int i = 0; i < 4; ++i) { v[i] = xr[tid + i * 256]; s1 += v[i]; s2 += v[i] * v[i]; }
  __shared__ float r1[256], r2[256];
  r1[tid] = s1; r2[tid] = s2; __syncthreads();
  for (int off = 128; off > 0; off >>= 1) {
    if (tid < off) { r1[tid] += r1[tid + off]; r2[tid] += r2[tid + off]; }
    __syncthreads();
  }
  float mean = r1[0] * (1.f / D_);
  float var  = r2[0] * (1.f / D_) - mean * mean;
  float rinv = rsqrtf(var + 1e-5f);
  for (int i = 0; i < 4; ++i) {
    int d = tid + i * 256;
    y[(size_t)row * D_ + d] = f2bf((v[i] - mean) * rinv * g[d] + bta[d]);
  }
}

// ---------------- WMMA bf16 GEMM (double-buffered, async LDS staging) ----------------
// C[M][N] (f32 and/or bf16) = act(A[M][K]_bf16 @ Bt[N][K]_bf16^T + bias) + resid
// 256 threads = 8 waves (2 in M x 4 in N); block tile 128x128, K-step 32.
#define BM 128
#define BN 128
#define BK 32
#define LDT 40   // LDS row stride in bf16 (80B, 16B-aligned, pads banks)

__global__ __launch_bounds__(256) void gemm_bf16_kernel(
    const bf16* __restrict__ A, const bf16* __restrict__ Bt,
    float* Cf, bf16* Cbf,
    const float* __restrict__ bias, const float* resid,
    int Mdim, int Ndim, int Kdim, int act) {
  __shared__ __align__(16) bf16 ldsA[2][BM * LDT];
  __shared__ __align__(16) bf16 ldsB[2][BN * LDT];

  const int tid  = threadIdx.x;
  const int lane = tid & 31;
  const int wave = tid >> 5;
  const int wm   = wave >> 2;   // 0..1  (64 rows each)
  const int wn   = wave & 3;    // 0..3  (32 cols each)
  const int m0 = blockIdx.y * BM;
  const int n0 = blockIdx.x * BN;

  const int lrow = tid >> 1;    // 0..127 tile row for staging
  const int lseg = tid & 1;     // 16-bf16 segment
  const bool bvalid = (n0 + lrow < Ndim);

  v8f acc[4][2];
  v8f zero = {0.f, 0.f, 0.f, 0.f, 0.f, 0.f, 0.f, 0.f};
  for (int i = 0; i < 4; ++i) for (int j = 0; j < 2; ++j) acc[i][j] = zero;

  const int ktiles = Kdim / BK;

  // stage one K-tile (A always in range; B guarded; stale LDS beyond N never stored)
  auto stage = [&](int kt, int buf) {
    const bf16* asrc = A + (size_t)(m0 + lrow) * Kdim + kt * BK + lseg * 16;
    bf16* adst = &ldsA[buf][lrow * LDT + lseg * 16];
    const bf16* bsrc = Bt + (size_t)(n0 + lrow) * Kdim + kt * BK + lseg * 16;
    bf16* bdst = &ldsB[buf][lrow * LDT + lseg * 16];
#if HAS_ASYNC
    __builtin_amdgcn_global_load_async_to_lds_b128(GPTR(asrc),     LPTR(adst),     0, 0);
    __builtin_amdgcn_global_load_async_to_lds_b128(GPTR(asrc + 8), LPTR(adst + 8), 0, 0);
    if (bvalid) {
      __builtin_amdgcn_global_load_async_to_lds_b128(GPTR(bsrc),     LPTR(bdst),     0, 0);
      __builtin_amdgcn_global_load_async_to_lds_b128(GPTR(bsrc + 8), LPTR(bdst + 8), 0, 0);
    }
#else
    {
      const uint4* s4 = (const uint4*)asrc;
      uint4* d4 = (uint4*)adst;
      d4[0] = s4[0]; d4[1] = s4[1];
    }
    if (bvalid) {
      const uint4* s4 = (const uint4*)bsrc;
      uint4* d4 = (uint4*)bdst;
      d4[0] = s4[0]; d4[1] = s4[1];
    }
#endif
  };

  stage(0, 0);
  wait_async0();
  __syncthreads();

  for (int kt = 0; kt < ktiles; ++kt) {
    const int cur = kt & 1;
    if (kt + 1 < ktiles) stage(kt + 1, cur ^ 1);   // async DMA overlaps WMMA below

    // A fragment (ISA 16-bit A 16x32): lanes 0-15 K{0..7,16..23}, lanes 16-31 K{8..15,24..31}
    const int r16 = lane & 15;
    const int akb = (lane < 16) ? 0 : 8;
    V16 afrag[4], bfrag[2];
    for (int i = 0; i < 4; ++i) {
      const bf16* p = &ldsA[cur][(wm * 64 + i * 16 + r16) * LDT + akb];
      afrag[i].h[0] = *(const v8bf*)(p);        // K kb..kb+7
      afrag[i].h[1] = *(const v8bf*)(p + 16);   // K kb+16..kb+23
    }
    // B fragment (32x16): lanes 0-15 K{0..7,16..23}, lanes 16-31 K{8..15,24..31}; LDS holds B^T
    const int bkb = (lane < 16) ? 0 : 8;
    for (int j = 0; j < 2; ++j) {
      const bf16* p = &ldsB[cur][(wn * 32 + j * 16 + r16) * LDT + bkb];
      bfrag[j].h[0] = *(const v8bf*)(p);
      bfrag[j].h[1] = *(const v8bf*)(p + 16);
    }
    for (int i = 0; i < 4; ++i)
      for (int j = 0; j < 2; ++j)
        acc[i][j] = __builtin_amdgcn_wmma_f32_16x16x32_bf16(
            false, afrag[i].v, false, bfrag[j].v, (short)0, acc[i][j], false, false);

    wait_async0();       // our next-tile async loads have landed
    __syncthreads();     // everyone done reading cur + writing next
  }

  // epilogue: C/D layout — VGPR r: lanes 0-15 (M=r, N=lane), lanes 16-31 (M=r+8, N=lane-16)
  const int c16 = lane & 15;
  const int rhi = (lane < 16) ? 0 : 8;
  for (int j = 0; j < 2; ++j) {
    const int col = n0 + wn * 32 + j * 16 + c16;
    if (col >= Ndim) continue;
    const float bv = bias ? bias[col] : 0.f;
    for (int i = 0; i < 4; ++i) {
      for (int r = 0; r < 8; ++r) {
        const int row = m0 + wm * 64 + i * 16 + r + rhi;
        float v = acc[i][j][r] + bv;
        if (act == 1) v = 0.5f * v * (1.f + erff(v * 0.70710678f));   // exact GELU
        size_t offn = (size_t)row * Ndim + col;
        if (resid) v += resid[offn];
        if (Cf)  Cf[offn]  = v;
        if (Cbf) Cbf[offn] = f2bf(v);
      }
    }
  }
}

// ---------------- RoPE (in-place on qkv f32 buffer, q at 0 / k at D) ----------------
__global__ void rope_kernel(float* qkv) {
  int t = blockIdx.x * blockDim.x + threadIdx.x;
  if (t >= M_ * H_ * 32) return;
  int which = blockIdx.y;                 // 0 = q, 1 = k
  int i = t & 31;
  int h = (t >> 5) & (H_ - 1);
  int rowbs = t >> 9;                     // b*S+s
  int s = rowbs & (S_ - 1);
  size_t base = (size_t)rowbs * (3 * D_) + (size_t)which * D_ + h * HD_ + 2 * i;
  float inv = powf(10000.f, -(float)(2 * i) / (float)HD_);
  float ang = (float)s * inv;
  float c = cosf(ang), sn = sinf(ang);
  float x0 = qkv[base], x1 = qkv[base + 1];
  qkv[base]     = x0 * c - x1 * sn;
  qkv[base + 1] = x1 * c + x0 * sn;
}

// ---------------- indexer scores: sc[b,t,s] = sum_h w_h * relu(q_th . k_s), causal ----------------
__global__ __launch_bounds__(256) void idx_scores_kernel(const float* __restrict__ q,
                                                         const float* __restrict__ k,
                                                         const float* __restrict__ w,
                                                         float* __restrict__ scores) {
  int b = blockIdx.z, t = blockIdx.y;
  int s = blockIdx.x * 256 + threadIdx.x;
  __shared__ float qs[IH_ * ID_];
  __shared__ float ws[IH_];
  int row = b * S_ + t;
  qs[threadIdx.x] = q[(size_t)row * (IH_ * ID_) + threadIdx.x];
  if (threadIdx.x < IH_) ws[threadIdx.x] = w[(size_t)row * IH_ + threadIdx.x];
  __syncthreads();
  float sc = NEGF;
  if (s <= t) {
    const float* kr = k + (size_t)(b * S_ + s) * ID_;
    float dots[IH_] = {0.f, 0.f, 0.f, 0.f};
    for (int d = 0; d < ID_; ++d) {
      float kv = kr[d];
      for (int hh = 0; hh < IH_; ++hh) dots[hh] += qs[hh * ID_ + d] * kv;
    }
    sc = 0.f;
    for (int hh = 0; hh < IH_; ++hh) sc += ws[hh] * fmaxf(dots[hh], 0.f);
  }
  scores[(size_t)row * S_ + s] = sc;
}

// ---------------- top-64 per row via iterative argmax ----------------
__global__ __launch_bounds__(256) void topk_kernel(const float* __restrict__ scores,
                                                   int* __restrict__ selidx,
                                                   int* __restrict__ selcnt) {
  int row = blockIdx.x, tid = threadIdx.x;
  __shared__ float sv[S_];
  __shared__ float rv[256];
  __shared__ int   ri[256];
  __shared__ int done_s, cnt_s;
  for (int ss = tid; ss < S_; ss += 256) sv[ss] = scores[(size_t)row * S_ + ss];
  if (tid == 0) { done_s = 0; cnt_s = 0; }
  __syncthreads();
  for (int it = 0; it < TOPK_; ++it) {
    float lv = NEGF; int li = -1;
    for (int ss = tid; ss < S_; ss += 256) { float x = sv[ss]; if (x > lv) { lv = x; li = ss; } }
    rv[tid] = lv; ri[tid] = li;
    __syncthreads();
    for (int off = 128; off > 0; off >>= 1) {
      if (tid < off && rv[tid + off] > rv[tid]) { rv[tid] = rv[tid + off]; ri[tid] = ri[tid + off]; }
      __syncthreads();
    }
    if (tid == 0) {
      if (rv[0] > -1e29f && ri[0] >= 0) {
        selidx[(size_t)row * TOPK_ + cnt_s] = ri[0];
        sv[ri[0]] = NEGF;
        cnt_s++;
      } else done_s = 1;
    }
    __syncthreads();
    if (done_s) break;
  }
  if (tid == 0) selcnt[row] = cnt_s;
}

// ---------------- sparse gathered attention: softmax over <=64 selected keys ----------------
__global__ __launch_bounds__(64) void attn_kernel(const float* __restrict__ qkv,
                                                  const int* __restrict__ selidx,
                                                  const int* __restrict__ selcnt,
                                                  bf16* __restrict__ o) {
  int t = blockIdx.x, h = blockIdx.y, b = blockIdx.z;
  int tid = threadIdx.x;
  int row = b * S_ + t;
  __shared__ float qs[HD_];
  __shared__ float p[TOPK_];
  __shared__ int si[TOPK_];
  int cnt = selcnt[row];
  qs[tid] = qkv[(size_t)row * (3 * D_) + h * HD_ + tid];
  if (tid < cnt) si[tid] = selidx[(size_t)row * TOPK_ + tid];
  __syncthreads();
  if (tid < cnt) {
    const float* kr = qkv + (size_t)(b * S_ + si[tid]) * (3 * D_) + D_ + h * HD_;
    float dot = 0.f;
    for (int d = 0; d < HD_; ++d) dot += qs[d] * kr[d];
    p[tid] = dot * 0.125f;                 // 1/sqrt(64)
  }
  __syncthreads();
  if (tid == 0) {
    float m = NEGF;
    for (int j = 0; j < cnt; ++j) m = fmaxf(m, p[j]);
    float den = 0.f;
    for (int j = 0; j < cnt; ++j) { float e = expf(p[j] - m); p[j] = e; den += e; }
    float rden = 1.f / den;
    for (int j = 0; j < cnt; ++j) p[j] *= rden;
  }
  __syncthreads();
  float accv = 0.f;
  for (int j = 0; j < cnt; ++j) {
    const float* vr = qkv + (size_t)(b * S_ + si[j]) * (3 * D_) + 2 * D_ + h * HD_;
    accv += p[j] * vr[tid];
  }
  o[(size_t)row * D_ + h * HD_ + tid] = f2bf(accv);
}

// ---------------- orchestration ----------------
extern "C" void kernel_launch(void* const* d_in, const int* in_sizes, int n_in,
                              void* d_out, int out_size, void* d_ws, size_t ws_size,
                              hipStream_t stream) {
  const int*   ids   = (const int*)  d_in[0];
  const float* tok   = (const float*)d_in[1];
  const float* pos   = (const float*)d_in[2];
  const float* ln1g  = (const float*)d_in[3];
  const float* ln1b  = (const float*)d_in[4];
  const float* qkvw  = (const float*)d_in[5];
  const float* wow   = (const float*)d_in[6];
  const float* idxqw = (const float*)d_in[7];
  const float* idxkw = (const float*)d_in[8];
  const float* idxww = (const float*)d_in[9];
  const float* ln2g  = (const float*)d_in[10];
  const float* ln2b  = (const float*)d_in[11];
  const float* w1    = (const float*)d_in[12];
  const float* b1    = (const float*)d_in[13];
  const float* w2    = (const float*)d_in[14];
  const float* b2    = (const float*)d_in[15];
  const float* lnfg  = (const float*)d_in[16];
  const float* lnfb  = (const float*)d_in[17];
  const float* headw = (const float*)d_in[18];
  float* out = (float*)d_out;

  char* ws = (char*)d_ws;
  size_t off = 0;
  auto alloc = [&](size_t bytes) -> char* {
    size_t a = (off + 255) & ~(size_t)255;
    off = a + bytes;
    return ws + a;
  };
  float* hbuf = (float*)alloc((size_t)M_ * D_ * 4);
  bf16*  nxbf = (bf16*) alloc((size_t)M_ * D_ * 2);
  float* qkvb = (float*)alloc((size_t)M_ * 3 * D_ * 4);
  bf16*  obf  = (bf16*) alloc((size_t)M_ * D_ * 2);
  float* iq   = (float*)alloc((size_t)M_ * IH_ * ID_ * 4);
  float* ik   = (float*)alloc((size_t)M_ * ID_ * 4);
  float* iw   = (float*)alloc((size_t)M_ * IH_ * 4);
  float* sc   = (float*)alloc((size_t)B_ * S_ * S_ * 4);
  int*   sidx = (int*)  alloc((size_t)M_ * TOPK_ * 4);
  int*   scnt = (int*)  alloc((size_t)M_ * 4);
  bf16*  f1bf = (bf16*) alloc((size_t)M_ * F_ * 2);
  bf16*  qkvT = (bf16*) alloc((size_t)L_ * 3 * D_ * D_ * 2);
  bf16*  woT  = (bf16*) alloc((size_t)L_ * D_ * D_ * 2);
  bf16*  iqT  = (bf16*) alloc((size_t)L_ * IH_ * ID_ * D_ * 2);
  bf16*  ikT  = (bf16*) alloc((size_t)L_ * ID_ * D_ * 2);
  bf16*  iwT  = (bf16*) alloc((size_t)L_ * IH_ * D_ * 2);
  bf16*  w1T  = (bf16*) alloc((size_t)L_ * F_ * D_ * 2);
  bf16*  w2T  = (bf16*) alloc((size_t)L_ * D_ * F_ * 2);
  bf16*  hdT  = (bf16*) alloc((size_t)V_ * D_ * 2);

  dim3 blk(256);
  auto cdiv = [](long a, long b) { return (unsigned)((a + b - 1) / b); };

  // one-shot weight conversions (deterministic; re-done every call)
  for (int l = 0; l < L_; ++l) {
    cvtT_kernel<<<cdiv((long)3 * D_ * D_, 256), blk, 0, stream>>>(qkvw + (size_t)l * D_ * 3 * D_, qkvT + (size_t)l * 3 * D_ * D_, D_, 3 * D_);
    cvtT_kernel<<<cdiv((long)D_ * D_, 256), blk, 0, stream>>>(wow + (size_t)l * D_ * D_, woT + (size_t)l * D_ * D_, D_, D_);
    cvtT_kernel<<<cdiv((long)IH_ * ID_ * D_, 256), blk, 0, stream>>>(idxqw + (size_t)l * D_ * IH_ * ID_, iqT + (size_t)l * IH_ * ID_ * D_, D_, IH_ * ID_);
    cvtT_kernel<<<cdiv((long)ID_ * D_, 256), blk, 0, stream>>>(idxkw + (size_t)l * D_ * ID_, ikT + (size_t)l * ID_ * D_, D_, ID_);
    cvtT_kernel<<<cdiv((long)IH_ * D_, 256), blk, 0, stream>>>(idxww + (size_t)l * D_ * IH_, iwT + (size_t)l * IH_ * D_, D_, IH_);
    cvtT_kernel<<<cdiv((long)F_ * D_, 256), blk, 0, stream>>>(w1 + (size_t)l * D_ * F_, w1T + (size_t)l * F_ * D_, D_, F_);
    cvtT_kernel<<<cdiv((long)D_ * F_, 256), blk, 0, stream>>>(w2 + (size_t)l * F_ * D_, w2T + (size_t)l * D_ * F_, F_, D_);
  }
  cvtT_kernel<<<cdiv((long)V_ * D_, 256), blk, 0, stream>>>(headw, hdT, D_, V_);

  embed_kernel<<<cdiv((long)M_ * D_, 256), blk, 0, stream>>>(ids, tok, pos, hbuf);

  for (int l = 0; l < L_; ++l) {
    ln_kernel<<<M_, blk, 0, stream>>>(hbuf, ln1g + l * D_, ln1b + l * D_, nxbf);
    gemm_bf16_kernel<<<dim3(3 * D_ / BN, M_ / BM), blk, 0, stream>>>(
        nxbf, qkvT + (size_t)l * 3 * D_ * D_, qkvb, nullptr, nullptr, nullptr, M_, 3 * D_, D_, 0);
    rope_kernel<<<dim3(cdiv((long)M_ * H_ * 32, 256), 2), blk, 0, stream>>>(qkvb);

    gemm_bf16_kernel<<<dim3(cdiv(IH_ * ID_, BN), M_ / BM), blk, 0, stream>>>(
        nxbf, iqT + (size_t)l * IH_ * ID_ * D_, iq, nullptr, nullptr, nullptr, M_, IH_ * ID_, D_, 0);
    gemm_bf16_kernel<<<dim3(1, M_ / BM), blk, 0, stream>>>(
        nxbf, ikT + (size_t)l * ID_ * D_, ik, nullptr, nullptr, nullptr, M_, ID_, D_, 0);
    gemm_bf16_kernel<<<dim3(1, M_ / BM), blk, 0, stream>>>(
        nxbf, iwT + (size_t)l * IH_ * D_, iw, nullptr, nullptr, nullptr, M_, IH_, D_, 0);

    idx_scores_kernel<<<dim3(S_ / 256, S_, B_), blk, 0, stream>>>(iq, ik, iw, sc);
    topk_kernel<<<M_, blk, 0, stream>>>(sc, sidx, scnt);
    attn_kernel<<<dim3(S_, H_, B_), dim3(64), 0, stream>>>(qkvb, sidx, scnt, obf);

    gemm_bf16_kernel<<<dim3(D_ / BN, M_ / BM), blk, 0, stream>>>(
        obf, woT + (size_t)l * D_ * D_, hbuf, nullptr, nullptr, hbuf, M_, D_, D_, 0);
    ln_kernel<<<M_, blk, 0, stream>>>(hbuf, ln2g + l * D_, ln2b + l * D_, nxbf);
    gemm_bf16_kernel<<<dim3(F_ / BN, M_ / BM), blk, 0, stream>>>(
        nxbf, w1T + (size_t)l * F_ * D_, nullptr, f1bf, b1 + (size_t)l * F_, nullptr, M_, F_, D_, 1);
    gemm_bf16_kernel<<<dim3(D_ / BN, M_ / BM), blk, 0, stream>>>(
        f1bf, w2T + (size_t)l * D_ * F_, hbuf, nullptr, b2 + (size_t)l * D_, hbuf, M_, D_, F_, 0);
  }

  ln_kernel<<<M_, blk, 0, stream>>>(hbuf, lnfg, lnfb, nxbf);
  gemm_bf16_kernel<<<dim3(V_ / BN, M_ / BM), blk, 0, stream>>>(
      nxbf, hdT, out, nullptr, nullptr, nullptr, M_, V_, D_, 0);
}
